// VulnerabilityDetection_84902913508090
// MI455X (gfx1250) — matrix operand order repaired and checked by reading, not compiled, not run on hardware
//
#include <hip/hip_runtime.h>
#include <hip/hip_bf16.h>

// ---------------------------------------------------------------------------
// CDNA5 / gfx1250 fused GAT+GCN head.  wave32, WMMA bf16 16x16x32.
// ---------------------------------------------------------------------------

#define NN    8192
#define HH    256
#define EE    262144
#define ALPHA 0.2f

typedef __bf16 bf16_t;
typedef bf16_t v16bf __attribute__((ext_vector_type(16)));
typedef float  v8f   __attribute__((ext_vector_type(8)));

struct alignas(16) Pack8 { bf16_t h[8]; };

__device__ __forceinline__ float lrelu(float x) { return x > 0.0f ? x : ALPHA * x; }
__device__ __forceinline__ float eluf(float x)  { return x > 0.0f ? x : (__expf(x) - 1.0f); }

__device__ __forceinline__ v8f wmma_bf16(const v16bf& a, const v16bf& b, const v8f& c) {
    return __builtin_amdgcn_wmma_f32_16x16x32_bf16(
        /*neg_a=*/false, a, /*neg_b=*/false, b,
        /*c_mod=*/(short)0, c, /*reuse_a=*/false, /*reuse_b=*/false);
}

// --- K0: W_attT[c][k] = bf16(W_att[k][c]) ----------------------------------
__global__ __launch_bounds__(256) void k_prep(const float* __restrict__ W,
                                              bf16_t* __restrict__ WT) {
    int t = blockIdx.x * 256 + threadIdx.x;   // 65536 threads
    int c = t >> 8, k = t & 255;
    WT[c * HH + k] = (bf16_t)W[k * HH + c];
}

// --- K init: deg=1 (self loop), cw=0, g=0 ----------------------------------
__global__ __launch_bounds__(256) void k_init(float* __restrict__ deg,
                                              float* __restrict__ cw,
                                              float* __restrict__ g) {
    int t = blockIdx.x * 256 + threadIdx.x;
    if (t < NN) { deg[t] = 1.0f; cw[t] = 0.0f; }
    if (t < HH) g[t] = 0.0f;
}

// --- K deg: in-degree via atomics ------------------------------------------
__global__ __launch_bounds__(256) void k_deg(const int* __restrict__ row,
                                             float* __restrict__ deg) {
    int e = blockIdx.x * 256 + threadIdx.x;
    if (e < EE) atomicAdd(&deg[row[e]], 1.0f);
}

// --- K dinv: dinv = rsqrt(deg); cw = dinv^2 (self-loop term) ---------------
__global__ __launch_bounds__(256) void k_dinv(const float* __restrict__ deg,
                                              float* __restrict__ dinv,
                                              float* __restrict__ cw) {
    int t = blockIdx.x * 256 + threadIdx.x;
    if (t < NN) { float di = rsqrtf(deg[t]); dinv[t] = di; cw[t] = di * di; }
}

// --- K cw: cw[col] += dinv[row]*dinv[col] over edges -----------------------
__global__ __launch_bounds__(256) void k_cw(const int* __restrict__ row,
                                            const int* __restrict__ col,
                                            const float* __restrict__ dinv,
                                            float* __restrict__ cw) {
    int e = blockIdx.x * 256 + threadIdx.x;
    if (e < EE) atomicAdd(&cw[col[e]], dinv[row[e]] * dinv[col[e]]);
}

// --- K1: Wh = feat @ W_att  (WMMA bf16), emit WhT bf16 + s,d vectors -------
// grid 512, block 256 (8 waves). Each wg: rows i0..i0+15, each wave: 32 cols.
__global__ __launch_bounds__(256) void k_gemm1(const float* __restrict__ feat,
                                               const bf16_t* __restrict__ WT,
                                               const float* __restrict__ a_src,
                                               const float* __restrict__ a_dst,
                                               bf16_t* __restrict__ whT,
                                               float* __restrict__ s,
                                               float* __restrict__ dvec) {
    const int tid  = threadIdx.x;
    const int lane = tid & 31;
    const int wave = tid >> 5;
    const int i0   = blockIdx.x * 16;
    const int r    = lane & 15;
    const int half = lane >> 4;
    const int kb   = half * 8;
    const int Mh   = half * 8;
    const int c0   = wave * 32;

    v8f acc[2] = {{}, {}};

    for (int kc = 0; kc < HH; kc += 32) {
        // A fragment: row i0+r, K = {half*16 .. half*16+15}
        const float* fr = feat + (i0 + r) * HH + kc + half * 16;
        float4 f0 = *(const float4*)(fr);
        float4 f1 = *(const float4*)(fr + 4);
        float4 f2 = *(const float4*)(fr + 8);
        float4 f3 = *(const float4*)(fr + 12);
        v16bf a;
        a[0]=(bf16_t)f0.x; a[1]=(bf16_t)f0.y; a[2]=(bf16_t)f0.z; a[3]=(bf16_t)f0.w;
        a[4]=(bf16_t)f1.x; a[5]=(bf16_t)f1.y; a[6]=(bf16_t)f1.z; a[7]=(bf16_t)f1.w;
        a[8]=(bf16_t)f2.x; a[9]=(bf16_t)f2.y; a[10]=(bf16_t)f2.z; a[11]=(bf16_t)f2.w;
        a[12]=(bf16_t)f3.x; a[13]=(bf16_t)f3.y; a[14]=(bf16_t)f3.z; a[15]=(bf16_t)f3.w;
        // B fragments: lane = column N, 16 contiguous K in WT
        v16bf b0 = *(const v16bf*)(WT + (c0      + r) * HH + kc + half * 16);
        v16bf b1 = *(const v16bf*)(WT + (c0 + 16 + r) * HH + kc + half * 16);
        acc[0] = wmma_bf16(a, b0, acc[0]);
        acc[1] = wmma_bf16(a, b1, acc[1]);
    }

    // store WhT (bf16, column-major over nodes): lane holds col c, M = Mh+v
    #pragma unroll
    for (int t = 0; t < 2; ++t) {
        int c = c0 + t * 16 + r;
        Pack8 p;
        #pragma unroll
        for (int v = 0; v < 8; ++v) p.h[v] = (bf16_t)acc[t][v];
        *(Pack8*)(whT + (size_t)c * NN + i0 + Mh) = p;
    }

    // s_i = Wh[i]·a_src, d_i = Wh[i]·a_dst : per-lane partials then reduce
    float as0 = a_src[c0 + r],      ad0 = a_dst[c0 + r];
    float as1 = a_src[c0 + 16 + r], ad1 = a_dst[c0 + 16 + r];
    float ps[8], pd[8];
    #pragma unroll
    for (int v = 0; v < 8; ++v) {
        ps[v] = acc[0][v] * as0 + acc[1][v] * as1;
        pd[v] = acc[0][v] * ad0 + acc[1][v] * ad1;
    }
    #pragma unroll
    for (int off = 1; off < 16; off <<= 1) {
        #pragma unroll
        for (int v = 0; v < 8; ++v) {
            ps[v] += __shfl_xor(ps[v], off);
            pd[v] += __shfl_xor(pd[v], off);
        }
    }
    __shared__ float ls[16], ld_[16];
    if (tid < 16) { ls[tid] = 0.0f; ld_[tid] = 0.0f; }
    __syncthreads();
    if ((lane & 15) == 0) {              // lanes 0 and 16 hold half-row sums
        #pragma unroll
        for (int v = 0; v < 8; ++v) {
            atomicAdd(&ls[Mh + v], ps[v]);
            atomicAdd(&ld_[Mh + v], pd[v]);
        }
    }
    __syncthreads();
    if (tid < 16) { s[i0 + tid] = ls[tid]; dvec[i0 + tid] = ld_[tid]; }
}

// --- K3: fused flash attention: h1 = elu(softmax(lrelu(s_i+d_j)) @ Wh), ----
//         then g[c] += sum_i cw[i]*h1[i][c]   (the whole GCN mean collapses)
// Cooperative P-tile production: each wave builds ONE 16x32 bf16 P fragment
// per 256-j superchunk, shares it via double-buffered LDS; every wave then
// consumes all 8 fragments -> 16 WMMAs per barrier, 8x less exp/VALU work.
__global__ __launch_bounds__(256) void k_attn(const float* __restrict__ s,
                                              const float* __restrict__ dvec,
                                              const bf16_t* __restrict__ whT,
                                              const float* __restrict__ cw,
                                              float* __restrict__ g) {
    __shared__ float sd[NN];                 // 32 KB: d staged once
    __shared__ v16bf afr[2][8][32];          // 16 KB: double-buffered A frags
    __shared__ float zrow[16];
    __shared__ float red[8];
    const int tid  = threadIdx.x;
    const int lane = tid & 31;
    const int wave = tid >> 5;
    const int i0   = blockIdx.x * 16;
    const int r    = lane & 15;
    const int half = lane >> 4;
    const int kb   = half * 8;
    const int Mh   = half * 8;
    const int c0   = wave * 32;

    // stage d into LDS, compute global max(d) (exact softmax max: lrelu monotone)
    float lm = -1e30f;
    for (int k = tid; k < NN; k += 256) { float v = dvec[k]; sd[k] = v; lm = fmaxf(lm, v); }
    #pragma unroll
    for (int off = 16; off; off >>= 1) lm = fmaxf(lm, __shfl_xor(lm, off));
    if (lane == 0) red[wave] = lm;
    if (tid < 16) zrow[tid] = 0.0f;
    __syncthreads();
    float dmax = red[0];
    #pragma unroll
    for (int w = 1; w < 8; ++w) dmax = fmaxf(dmax, red[w]);

    const float sr = s[i0 + r];
    const float m  = lrelu(sr + dmax);

    const bf16_t* bb0 = whT + (size_t)(c0 + r) * NN + half * 16;
    const bf16_t* bb1 = whT + (size_t)(c0 + 16 + r) * NN + half * 16;

    v8f acc0 = {}, acc1 = {};
    float z = 0.0f;

    for (int jj = 0; jj < NN; jj += 256) {       // 32 superchunks
        const int pb = (jj >> 8) & 1;
        const int j0 = jj + wave * 32;           // this wave's P chunk
        // ---- produce: P fragment for my chunk -------------------------
        float4 d0 = *(const float4*)(sd + j0 + half * 16);
        float4 d1 = *(const float4*)(sd + j0 + half * 16 + 4);
        float4 d2 = *(const float4*)(sd + j0 + half * 16 + 8);
        float4 d3 = *(const float4*)(sd + j0 + half * 16 + 12);
        float p[16];
        p[0]  = __expf(lrelu(sr + d0.x) - m);  p[1]  = __expf(lrelu(sr + d0.y) - m);
        p[2]  = __expf(lrelu(sr + d0.z) - m);  p[3]  = __expf(lrelu(sr + d0.w) - m);
        p[4]  = __expf(lrelu(sr + d1.x) - m);  p[5]  = __expf(lrelu(sr + d1.y) - m);
        p[6]  = __expf(lrelu(sr + d1.z) - m);  p[7]  = __expf(lrelu(sr + d1.w) - m);
        p[8]  = __expf(lrelu(sr + d2.x) - m);  p[9]  = __expf(lrelu(sr + d2.y) - m);
        p[10] = __expf(lrelu(sr + d2.z) - m);  p[11] = __expf(lrelu(sr + d2.w) - m);
        p[12] = __expf(lrelu(sr + d3.x) - m);  p[13] = __expf(lrelu(sr + d3.y) - m);
        p[14] = __expf(lrelu(sr + d3.z) - m);  p[15] = __expf(lrelu(sr + d3.w) - m);
        v16bf a;
        #pragma unroll
        for (int i = 0; i < 16; ++i) { z += p[i]; a[i] = (bf16_t)p[i]; }
        afr[pb][wave][lane] = a;
        __syncthreads();
        // ---- consume: all 8 fragments vs my 2 column tiles ------------
        #pragma unroll
        for (int q = 0; q < 8; ++q) {
            v16bf aq = afr[pb][q][lane];
            v16bf b0 = *(const v16bf*)(bb0 + jj + q * 32);
            v16bf b1 = *(const v16bf*)(bb1 + jj + q * 32);
            acc0 = wmma_bf16(aq, b0, acc0);
            acc1 = wmma_bf16(aq, b1, acc1);
        }
    }

    // per-row partition function: this wave saw 1/8 of j; rows r on lanes r,r+16
    z += __shfl_xor(z, 16);
    if (lane < 16) atomicAdd(&zrow[lane], z);
    __syncthreads();

    float4 ca = *(const float4*)(cw + i0 + Mh);
    float4 cb = *(const float4*)(cw + i0 + Mh + 4);
    float cwv[8] = {ca.x, ca.y, ca.z, ca.w, cb.x, cb.y, cb.z, cb.w};

    float part0 = 0.0f, part1 = 0.0f;
    #pragma unroll
    for (int v = 0; v < 8; ++v) {
        float zv = zrow[Mh + v];                     // Z of row Mh+v
        float h0 = eluf(acc0[v] / zv);
        float h1 = eluf(acc1[v] / zv);
        part0 += cwv[v] * h0;
        part1 += cwv[v] * h1;
    }
    part0 += __shfl_xor(part0, 16);
    part1 += __shfl_xor(part1, 16);
    if (lane < 16) {
        atomicAdd(&g[c0 + lane],      part0);
        atomicAdd(&g[c0 + 16 + lane], part1);
    }
}

// --- K4: out = softmax( (g/N + gcn_b) @ out_w ... @ gcn_w folded ) ---------
__global__ __launch_bounds__(256) void k_final(const float* __restrict__ g,
                                               const float* __restrict__ gcn_w,
                                               const float* __restrict__ gcn_b,
                                               const float* __restrict__ out_w,
                                               const float* __restrict__ out_b,
                                               float* __restrict__ out) {
    __shared__ float gs[HH], r0[HH], r1[HH];
    int t = threadIdx.x;
    gs[t] = g[t];
    __syncthreads();
    float u = 0.0f;
    for (int k = 0; k < HH; ++k) u += gs[k] * gcn_w[k * HH + t];
    float v = u * (1.0f / (float)NN) + gcn_b[t];
    r0[t] = v * out_w[t * 2 + 0];
    r1[t] = v * out_w[t * 2 + 1];
    __syncthreads();
    for (int off = 128; off; off >>= 1) {
        if (t < off) { r0[t] += r0[t + off]; r1[t] += r1[t + off]; }
        __syncthreads();
    }
    if (t == 0) {
        float a = r0[0] + out_b[0], b = r1[0] + out_b[1];
        float mx = fmaxf(a, b);
        float ea = __expf(a - mx), eb = __expf(b - mx);
        float inv = 1.0f / (ea + eb);
        out[0] = ea * inv;
        out[1] = eb * inv;
    }
}

// ---------------------------------------------------------------------------
extern "C" void kernel_launch(void* const* d_in, const int* in_sizes, int n_in,
                              void* d_out, int out_size, void* d_ws, size_t ws_size,
                              hipStream_t stream) {
    const float* feat  = (const float*)d_in[0];
    const int*   eidx  = (const int*)  d_in[1];   // [2,E] int32: row then col
    const float* W_att = (const float*)d_in[2];
    const float* a_src = (const float*)d_in[3];
    const float* a_dst = (const float*)d_in[4];
    const float* gcn_w = (const float*)d_in[5];
    const float* gcn_b = (const float*)d_in[6];
    const float* out_w = (const float*)d_in[7];
    const float* out_b = (const float*)d_in[8];
    float* out = (float*)d_out;

    char* ws = (char*)d_ws;
    bf16_t* WT  = (bf16_t*)ws;                              // 256*256 bf16 = 128 KB
    bf16_t* whT = (bf16_t*)(ws + 131072);                   // 256*8192 bf16 = 4 MB
    float*  s    = (float*)(ws + 131072 + 4194304);
    float*  dv   = s    + NN;
    float*  deg  = dv   + NN;
    float*  dinv = deg  + NN;
    float*  cw   = dinv + NN;
    float*  g    = cw   + NN;

    const int* row = eidx;        // destination
    const int* col = eidx + EE;   // source

    k_prep <<<256, 256, 0, stream>>>(W_att, WT);
    k_init <<<32, 256, 0, stream>>>(deg, cw, g);
    k_gemm1<<<512, 256, 0, stream>>>(feat, WT, a_src, a_dst, whT, s, dv);
    k_deg  <<<EE / 256, 256, 0, stream>>>(row, deg);
    k_dinv <<<32, 256, 0, stream>>>(deg, dinv, cw);
    k_cw   <<<EE / 256, 256, 0, stream>>>(row, col, dinv, cw);
    k_attn <<<512, 256, 0, stream>>>(s, dv, whT, cw, g);
    k_final<<<1, 256, 0, stream>>>(g, gcn_w, gcn_b, out_w, out_b, out);
}
